// PointCloudTCN_403726926231
// MI455X (gfx1250) — matrix operand order repaired and checked by reading, not compiled, not run on hardware
//
#include <hip/hip_runtime.h>
#include <math.h>

// ---------------------------------------------------------------------------
// PointCloudTCN for MI455X (gfx1250, wave32, WMMA + TDM).
//   - kNN distance GEMM on v_wmma_f32_16x16x32_f16 (dominant FLOPs),
//     branch-free register top-3; source features staged into LDS via the
//     Tensor Data Mover (double buffered, one DMA per workgroup chunk)
//   - head MLPs (10->100->100->100->{5,1}) as WMMA GEMMs, fused epilogue
//   - small per-edge MLPs as register-streamed scalar kernels
// ---------------------------------------------------------------------------

#define N_NODES 16384
#define NTILES  (N_NODES / 16)     // 1024

// kNN LDS staging: 256 source nodes (64B rows) per chunk, double buffered.
#define CH_ROWS   256
#define CH_HALVES (CH_ROWS * 32)   // halves per chunk buffer
#define CH_BYTES  (CH_ROWS * 64)   // 16384 bytes
#define NCHUNK    (N_NODES / CH_ROWS)  // 64
#define TILES_PER_CH (CH_ROWS / 16)    // 16

typedef _Float16 half_t;
typedef __attribute__((ext_vector_type(16))) _Float16 v16h;
typedef __attribute__((ext_vector_type(8)))  float    v8f;
typedef __attribute__((ext_vector_type(4)))  unsigned int v4u;
typedef __attribute__((ext_vector_type(8)))  int      v8i;
typedef __attribute__((ext_vector_type(4)))  int      v4i;

union HFrag { v16h v; uint4 u[2]; };

#if defined(__has_builtin)
#  if __has_builtin(__builtin_amdgcn_tensor_load_to_lds)
#    define HAS_TDM 1
#  endif
#endif
#ifndef HAS_TDM
#  define HAS_TDM 0
#endif

// ---------------------------------------------------------------------------
// Feature conversion for the kNN WMMA trick:
//   A[k] = -2*x[k] (k<d), A[10] = |x|^2     (source operand)
//   B[k] =    x[k] (k<d), B[10] = 1         (target operand)
// => WMMA D = |x_s|^2 - 2 x_s.x_t  (ordering-equivalent to squared distance)
// Rows are 32 halves (64B) zero padded; K=32 for v_wmma_f32_16x16x32_f16.
// ---------------------------------------------------------------------------
__global__ void cvt_feat_kernel(const float* __restrict__ hin, int d,
                                half_t* __restrict__ A, half_t* __restrict__ B) {
  int i = blockIdx.x * blockDim.x + threadIdx.x;
  if (i >= N_NODES) return;
  float x[10];
  float sq = 0.f;
#pragma unroll
  for (int t = 0; t < 10; ++t) {
    float v = (t < d) ? hin[(size_t)i * d + t] : 0.f;
    x[t] = v; sq += v * v;
  }
  half_t* ar = A + (size_t)i * 32;
  half_t* br = B + (size_t)i * 32;
#pragma unroll
  for (int t = 0; t < 32; ++t) {
    float av = 0.f, bv = 0.f;
    if (t < 10)       { av = -2.f * x[t]; bv = x[t]; }
    else if (t == 10) { av = sq;          bv = 1.f;  }
    ar[t] = (half_t)av;
    br[t] = (half_t)bv;
  }
}

// Branch-free sorted top-3 (min) insertion: pure v_cmp/v_cndmask, no EXEC
// manipulation, no scratch. Updates must read pre-update neighbors, so order
// is d2 (from old d1), d1 (from old d0), d0.
__device__ __forceinline__ void top3_insert(float v, int id,
                                            float& d0, float& d1, float& d2,
                                            int& i0, int& i1, int& i2) {
  const bool lt0 = v < d0;
  const bool lt1 = v < d1;
  const bool lt2 = v < d2;
  d2 = lt1 ? d1 : (lt2 ? v : d2);
  i2 = lt1 ? i1 : (lt2 ? id : i2);
  d1 = lt0 ? d0 : (lt1 ? v : d1);
  i1 = lt0 ? i0 : (lt1 ? id : i1);
  d0 = lt0 ? v : d0;
  i0 = lt0 ? id : i0;
}

// One TDM descriptor: contiguous CH_BYTES from global -> LDS buffer `parity`.
// D# per CDNA5 ISA ch.8: group0 {count=1, lds_addr, global_addr, type=2},
// group1 {data_size=2B, tensor_dim0=tile_dim0=CH_BYTES/2, 1 row}.
__device__ __forceinline__ void stage_chunk(const half_t* __restrict__ gbase,
                                            int chunk, int parity,
                                            half_t* __restrict__ smem) {
#if HAS_TDM
  const unsigned long long ga =
      (unsigned long long)gbase + (unsigned long long)chunk * CH_BYTES;
  v4u g0;
  g0.x = 1u;                                    // count=1 (valid user D#)
  g0.y = (unsigned)(parity * CH_BYTES);         // LDS byte address (static LDS base 0)
  g0.z = (unsigned)(ga & 0xffffffffu);          // global_addr[31:0]
  g0.w = (unsigned)((ga >> 32) & 0x1ffffffu) | (2u << 30);  // addr[56:32] | type=2
  v8i g1;
  g1[0] = (1 << 16);                 // workgroup_mask=0, data_size=1 (2 bytes)
  g1[1] = (int)((CH_BYTES / 2) << 16);  // tensor_dim0 low16 (8192)
  g1[2] = (1 << 16);                 // tensor_dim0 hi=0 | tensor_dim1=1
  g1[3] = (int)((CH_BYTES / 2) << 16);  // tensor_dim1 hi=0 | tile_dim0=8192
  g1[4] = 1;                         // tile_dim1=1, tile_dim2=0
  g1[5] = (int)(CH_BYTES / 2);       // tensor_dim0_stride low32
  g1[6] = 0;
  g1[7] = 0;
  const v4i z4 = {0, 0, 0, 0};
#if __clang_major__ >= 23
  const v8i z8 = {0, 0, 0, 0, 0, 0, 0, 0};
  __builtin_amdgcn_tensor_load_to_lds(g0, g1, z4, z4, z8, 0);
#else
  __builtin_amdgcn_tensor_load_to_lds(g0, g1, z4, z4, 0);
#endif
  (void)smem;
#else
  // Fallback: wave-cooperative copy (issuing wave only).
  const int lane = threadIdx.x & 31;
  const uint4* g = (const uint4*)(gbase + (size_t)chunk * CH_HALVES);
  uint4* l = (uint4*)(smem + (size_t)parity * CH_HALVES);
  for (int t = lane; t < CH_BYTES / 16; t += 32) l[t] = g[t];
#endif
}

__device__ __forceinline__ void stage_wait() {
#if HAS_TDM
  __builtin_amdgcn_s_wait_tensorcnt(0);
#endif
}

// ---------------------------------------------------------------------------
// kNN: one wave = 16 target nodes; block = 4 waves sharing the LDS-staged
// source stream. Wave 0 drives the TDM double buffer; every wave runs one
// v_wmma_f32_16x16x32_f16 per 16x16 tile and keeps a register top-3.
// Final cross-half merge via wave32 shuffles; lanes 0..15 write indices.
// ---------------------------------------------------------------------------
__global__ __launch_bounds__(128)
void knn_wmma_kernel(const half_t* __restrict__ Af, const half_t* __restrict__ Bf,
                     int* __restrict__ idxout) {
  __shared__ __align__(16) half_t smem[2 * CH_HALVES];   // 32 KB, double buffer

  const int  lane = threadIdx.x & 31;
  const int  wave = threadIdx.x >> 5;
  const int  col  = lane & 15;
  const bool hiL  = (lane >= 16);
  const int  tgtBase = (blockIdx.x * 4 + wave) * 16;
  const int  tgtIdx  = tgtBase + col;

  // B fragment (16x16x32 f16 B operand): lanes<16 hold K {0..7,16..23} of column `col`,
  // lanes>=16 hold K {8..15,24..31}.
  HFrag b;
  {
    const half_t* br = Bf + (size_t)tgtIdx * 32 + (hiL ? 8 : 0);
    b.u[0] = *(const uint4*)(br);
    b.u[1] = *(const uint4*)(br + 16);
  }

  float d0 = 3.0e38f, d1 = 3.0e38f, d2 = 3.0e38f;
  int   i0 = 0, i1 = 1, i2 = 2;

  // Prologue: DMA chunk 0 into buffer 0.
  if (wave == 0) stage_chunk(Af, 0, 0, smem);

  for (int ch = 0; ch < NCHUNK; ++ch) {
    const int par = ch & 1;
    if (wave == 0) stage_wait();       // chunk `ch` landed in LDS
    __syncthreads();                   // ...and is visible to all waves
    if (wave == 0 && ch + 1 < NCHUNK)  // prior barrier freed the other buffer
      stage_chunk(Af, ch + 1, par ^ 1, smem);

    const half_t* base = smem + (size_t)par * CH_HALVES;
#pragma unroll 4
    for (int t = 0; t < TILES_PER_CH; ++t) {
      // A fragment from LDS: lanes<16 read K {0..7,16..23} of row `col`,
      // lanes>=16 read K {8..15,24..31}.
      HFrag a;
      const half_t* lr = base + (size_t)(t * 16 + col) * 32 + (hiL ? 8 : 0);
      a.u[0] = *(const uint4*)(lr);
      a.u[1] = *(const uint4*)(lr + 16);

      v8f c = {};
      c = __builtin_amdgcn_wmma_f32_16x16x32_f16(false, a.v, false, b.v,
                                                 (short)0, c, false, false);

      const int mbase = (ch * TILES_PER_CH + t) * 16 + (hiL ? 8 : 0);
#pragma unroll
      for (int r = 0; r < 8; ++r) {
        const int id = mbase + r;
        const float v = (id == tgtIdx) ? 3.0e38f : c[r];  // mask self loop
        top3_insert(v, id, d0, d1, d2, i0, i1, i2);
      }
    }
    __syncthreads();   // everyone done reading `par` before it is re-filled
  }

  // Target t's candidates split between lane t and lane t+16.
  const float e0 = __shfl(d0, lane + 16, 32);
  const float e1 = __shfl(d1, lane + 16, 32);
  const float e2 = __shfl(d2, lane + 16, 32);
  const int   j0 = __shfl(i0, lane + 16, 32);
  const int   j1 = __shfl(i1, lane + 16, 32);
  const int   j2 = __shfl(i2, lane + 16, 32);
  top3_insert(e0, j0, d0, d1, d2, i0, i1, i2);
  top3_insert(e1, j1, d0, d1, d2, i0, i1, i2);
  top3_insert(e2, j2, d0, d1, d2, i0, i1, i2);

  if (!hiL) {
    idxout[(size_t)tgtIdx * 3 + 0] = i0;
    idxout[(size_t)tgtIdx * 3 + 1] = i1;
    idxout[(size_t)tgtIdx * 3 + 2] = i2;
  }
}

// ---------------------------------------------------------------------------
// Streamed 2-layer MLP (IN -> 100 relu -> 10, no final act). Weights are read
// with wave-uniform addresses (scalar loads); activations stay in registers.
// ---------------------------------------------------------------------------
template <int IN>
__device__ __forceinline__ void mlp_100_10(const float* __restrict__ in,
                                           const float* __restrict__ W1,
                                           const float* __restrict__ B1,
                                           const float* __restrict__ W2,
                                           const float* __restrict__ B2,
                                           float* __restrict__ out) {
#pragma unroll
  for (int o = 0; o < 10; ++o) out[o] = B2[o];
  for (int hh = 0; hh < 100; ++hh) {
    float t = B1[hh];
#pragma unroll
    for (int q = 0; q < IN; ++q) t = fmaf(in[q], W1[q * 100 + hh], t);
    t = fmaxf(t, 0.f);
#pragma unroll
    for (int o = 0; o < 10; ++o) out[o] = fmaf(t, W2[hh * 10 + o], out[o]);
  }
}

// DynamicEdgeConv: h_i = relu( sum_n MLP(cat[x_i, x_j - x_i]) )
template <int D>
__global__ void node_encoder_kernel(const float* __restrict__ hin,
                                    const int* __restrict__ idx, int k,
                                    const float* __restrict__ W1, const float* __restrict__ B1,
                                    const float* __restrict__ W2, const float* __restrict__ B2,
                                    float* __restrict__ hout) {
  int i = blockIdx.x * blockDim.x + threadIdx.x;
  if (i >= N_NODES) return;
  float xi[D];
#pragma unroll
  for (int t = 0; t < D; ++t) xi[t] = hin[(size_t)i * D + t];
  float acc[10];
#pragma unroll
  for (int o = 0; o < 10; ++o) acc[o] = 0.f;
  for (int n = 0; n < k; ++n) {
    int j = idx[(size_t)i * 3 + n];
    float in[2 * D];
#pragma unroll
    for (int t = 0; t < D; ++t) {
      in[t] = xi[t];
      in[D + t] = hin[(size_t)j * D + t] - xi[t];
    }
    float m[10];
    mlp_100_10<2 * D>(in, W1, B1, W2, B2, m);
#pragma unroll
    for (int o = 0; o < 10; ++o) acc[o] += m[o];
  }
#pragma unroll
  for (int o = 0; o < 10; ++o) hout[(size_t)i * 10 + o] = fmaxf(acc[o], 0.f);
}

// e = relu(MLP(cat[h_src, h_dst]))
__global__ void edge_encoder_kernel(const float* __restrict__ h,
                                    const int* __restrict__ idx, int k,
                                    const float* __restrict__ W1, const float* __restrict__ B1,
                                    const float* __restrict__ W2, const float* __restrict__ B2,
                                    float* __restrict__ e) {
  int t0 = blockIdx.x * blockDim.x + threadIdx.x;
  int i = t0 / 3, n = t0 % 3;
  if (i >= N_NODES || n >= k) return;
  int j = idx[(size_t)i * 3 + n];
  float in[20];
#pragma unroll
  for (int t = 0; t < 10; ++t) {
    in[t]      = h[(size_t)j * 10 + t];   // h_src
    in[10 + t] = h[(size_t)i * 10 + t];   // h_dst
  }
  float m[10];
  mlp_100_10<20>(in, W1, B1, W2, B2, m);
#pragma unroll
  for (int o = 0; o < 10; ++o) e[(size_t)(i * 3 + n) * 10 + o] = fmaxf(m[o], 0.f);
}

// e = MLP(cat[h_dst, h_src, e])   (in-place per edge, no final act)
__global__ void rel_kernel(const float* __restrict__ h, const int* __restrict__ idx, int k,
                           const float* __restrict__ W1, const float* __restrict__ B1,
                           const float* __restrict__ W2, const float* __restrict__ B2,
                           float* __restrict__ e) {
  int t0 = blockIdx.x * blockDim.x + threadIdx.x;
  int i = t0 / 3, n = t0 % 3;
  if (i >= N_NODES || n >= k) return;
  int j = idx[(size_t)i * 3 + n];
  float in[30];
#pragma unroll
  for (int t = 0; t < 10; ++t) {
    in[t]      = h[(size_t)i * 10 + t];                 // h_dst
    in[10 + t] = h[(size_t)j * 10 + t];                 // h_src
    in[20 + t] = e[(size_t)(i * 3 + n) * 10 + t];       // e
  }
  float m[10];
  mlp_100_10<30>(in, W1, B1, W2, B2, m);
#pragma unroll
  for (int o = 0; o < 10; ++o) e[(size_t)(i * 3 + n) * 10 + o] = m[o];
}

// h = 0.5*h + 0.5*MLP(cat[h, sum_k e])   (in-place per node)
__global__ void obj_kernel(float* __restrict__ h, const float* __restrict__ e, int k,
                           const float* __restrict__ W1, const float* __restrict__ B1,
                           const float* __restrict__ W2, const float* __restrict__ B2) {
  int i = blockIdx.x * blockDim.x + threadIdx.x;
  if (i >= N_NODES) return;
  float in[20];
#pragma unroll
  for (int t = 0; t < 10; ++t) in[t] = h[(size_t)i * 10 + t];
#pragma unroll
  for (int t = 0; t < 10; ++t) {
    float s = 0.f;
    for (int n = 0; n < k; ++n) s += e[(size_t)(i * 3 + n) * 10 + t];
    in[10 + t] = s;
  }
  float m[10];
  mlp_100_10<20>(in, W1, B1, W2, B2, m);
#pragma unroll
  for (int o = 0; o < 10; ++o) h[(size_t)i * 10 + o] = 0.5f * in[o] + 0.5f * m[o];
}

// ---------------------------------------------------------------------------
// Head MLPs as WMMA GEMMs: act f16 [N x 128] (K padded), weights transposed
// and converted to f16 [128 x 128] per layer.
// ---------------------------------------------------------------------------
__global__ void pack_act_kernel(const float* __restrict__ h, half_t* __restrict__ dst) {
  int t = blockIdx.x * blockDim.x + threadIdx.x;     // N*128 threads
  if (t >= N_NODES * 128) return;
  int i = t >> 7, c = t & 127;
  dst[t] = (half_t)((c < 10) ? h[(size_t)i * 10 + c] : 0.f);
}

// dst[n*128 + k] = W[k*Nout + n]  (zero padded)
__global__ void wt_pack_kernel(const float* __restrict__ W, int K, int Nout,
                               half_t* __restrict__ dst) {
  int t = blockIdx.x * blockDim.x + threadIdx.x;     // 128*128 threads
  if (t >= 128 * 128) return;
  int n = t >> 7, kk = t & 127;
  dst[t] = (half_t)((n < Nout && kk < K) ? W[(size_t)kk * Nout + n] : 0.f);
}

// mode: 0 = relu -> f16 activation buffer; 1 = linear -> f32; 2 = sigmoid+eps -> f32
__global__ __launch_bounds__(128)
void gemm16_kernel(const half_t* __restrict__ act, const half_t* __restrict__ wt,
                   const float* __restrict__ bias, int nout, int ntiles, int ksteps,
                   int mode, half_t* __restrict__ out16, float* __restrict__ outf) {
  const int  lane = threadIdx.x & 31;
  const int  wave = threadIdx.x >> 5;
  const int  tile = blockIdx.x * 4 + wave;
  const int  mt = tile / ntiles;
  const int  nt = tile - mt * ntiles;
  if (mt >= NTILES) return;
  const int  col = lane & 15;
  const bool hiL = (lane >= 16);
  const int  arow = mt * 16 + col;
  const int  n = nt * 16 + col;

  v8f c = {};
  for (int kk = 0; kk < ksteps; ++kk) {
    HFrag a, b;
    const half_t* ap = act + (size_t)arow * 128 + kk * 32 + (hiL ? 8 : 0);
    a.u[0] = *(const uint4*)(ap);
    a.u[1] = *(const uint4*)(ap + 16);
    const half_t* bp = wt + (size_t)n * 128 + kk * 32 + (hiL ? 8 : 0);
    b.u[0] = *(const uint4*)(bp);
    b.u[1] = *(const uint4*)(bp + 16);
    c = __builtin_amdgcn_wmma_f32_16x16x32_f16(false, a.v, false, b.v,
                                               (short)0, c, false, false);
  }

  const float bv = (n < nout) ? bias[n] : 0.f;
#pragma unroll
  for (int r = 0; r < 8; ++r) {
    const int m = mt * 16 + r + (hiL ? 8 : 0);
    float v = c[r] + bv;
    if (mode == 0) {
      out16[(size_t)m * 128 + n] = (half_t)fmaxf(v, 0.f);     // full 128-wide write keeps pad zero
    } else if (n < nout) {
      if (mode == 2) v = 1.f / (1.f + __expf(-v)) + 1e-11f;
      outf[(size_t)m * nout + n] = v;
    }
  }
}

// ---------------------------------------------------------------------------
// Host side
// ---------------------------------------------------------------------------
extern "C" void kernel_launch(void* const* d_in, const int* in_sizes, int n_in,
                              void* d_out, int out_size, void* d_ws, size_t ws_size,
                              hipStream_t stream) {
  (void)out_size; (void)ws_size;

  // Locate x (16384*6 floats) — either first (insertion order) or last
  // (jax sorted-key flatten puts "params" before "x").
  const bool x_first = (in_sizes[0] == N_NODES * 6);
  const float* x = (const float*)(x_first ? d_in[0] : d_in[n_in - 1]);
  int p = x_first ? 1 : 0;
  auto NXT = [&]() { return (const float*)d_in[p++]; };

  // Param leaves in jax sorted-dict-key order: B, X, blocks[.]{edge_encoder,
  // ins[.]{obj,rel}, node_encoder}, each MLP as [(W,b),...].
  const float *Bw[4], *Bb[4], *Xw[4], *Xb[4];
  for (int l = 0; l < 4; ++l) { Bw[l] = NXT(); Bb[l] = NXT(); }
  for (int l = 0; l < 4; ++l) { Xw[l] = NXT(); Xb[l] = NXT(); }
  struct BlkP {
    const float *eeW1, *eeB1, *eeW2, *eeB2;
    const float *objW1[3], *objB1[3], *objW2[3], *objB2[3];
    const float *relW1[3], *relB1[3], *relW2[3], *relB2[3];
    const float *neW1, *neB1, *neW2, *neB2;
  } blk[4];
  for (int b = 0; b < 4; ++b) {
    blk[b].eeW1 = NXT(); blk[b].eeB1 = NXT(); blk[b].eeW2 = NXT(); blk[b].eeB2 = NXT();
    for (int l = 0; l < 3; ++l) {
      blk[b].objW1[l] = NXT(); blk[b].objB1[l] = NXT();
      blk[b].objW2[l] = NXT(); blk[b].objB2[l] = NXT();
      blk[b].relW1[l] = NXT(); blk[b].relB1[l] = NXT();
      blk[b].relW2[l] = NXT(); blk[b].relB2[l] = NXT();
    }
    blk[b].neW1 = NXT(); blk[b].neB1 = NXT(); blk[b].neW2 = NXT(); blk[b].neB2 = NXT();
  }

  // Workspace layout (bytes)
  char* ws = (char*)d_ws;
  float*  hA    = (float*) (ws + 0);          // N*10 f32
  float*  hB    = (float*) (ws + 655360);     // N*10 f32
  float*  ebuf  = (float*) (ws + 1310720);    // N*3*10 f32
  int*    idxb  = (int*)   (ws + 3276800);    // N*3 i32
  half_t* Af    = (half_t*)(ws + 3473408);    // N*32 f16
  half_t* Bf    = (half_t*)(ws + 4521984);    // N*32 f16
  half_t* actA  = (half_t*)(ws + 5570560);    // N*128 f16
  half_t* actB  = (half_t*)(ws + 9764864);    // N*128 f16
  half_t* wslot = (half_t*)(ws + 13959168);   // 8 x (128*128) f16

  static const int KS[4] = {3, 3, 2, 1};
  const float* hin = x;
  float* hcur = hA;
  float* hoth = hB;

  for (int b = 0; b < 4; ++b) {
    const int k = KS[b];
    const int d = (b == 0) ? 6 : 10;
    cvt_feat_kernel<<<N_NODES / 256, 256, 0, stream>>>(hin, d, Af, Bf);
    knn_wmma_kernel<<<NTILES / 4, 128, 0, stream>>>(Af, Bf, idxb);
    if (b == 0)
      node_encoder_kernel<6><<<N_NODES / 256, 256, 0, stream>>>(
          hin, idxb, k, blk[b].neW1, blk[b].neB1, blk[b].neW2, blk[b].neB2, hcur);
    else
      node_encoder_kernel<10><<<N_NODES / 256, 256, 0, stream>>>(
          hin, idxb, k, blk[b].neW1, blk[b].neB1, blk[b].neW2, blk[b].neB2, hcur);
    edge_encoder_kernel<<<(N_NODES * 3) / 256, 256, 0, stream>>>(
        hcur, idxb, k, blk[b].eeW1, blk[b].eeB1, blk[b].eeW2, blk[b].eeB2, ebuf);
    for (int l = 0; l < 3; ++l) {
      rel_kernel<<<(N_NODES * 3) / 256, 256, 0, stream>>>(
          hcur, idxb, k, blk[b].relW1[l], blk[b].relB1[l], blk[b].relW2[l], blk[b].relB2[l], ebuf);
      obj_kernel<<<N_NODES / 256, 256, 0, stream>>>(
          hcur, ebuf, k, blk[b].objW1[l], blk[b].objB1[l], blk[b].objW2[l], blk[b].objB2[l]);
    }
    hin = hcur;
    float* tmp = hcur; hcur = hoth; hoth = tmp;
  }

  // Heads: X (10->100->100->100->5), B (10->100->100->100->1 + sigmoid + eps)
  static const int Kd[4] = {10, 100, 100, 100};
  for (int l = 0; l < 4; ++l) {
    wt_pack_kernel<<<64, 256, 0, stream>>>(Xw[l], Kd[l], (l == 3) ? 5 : 100,
                                           wslot + (size_t)l * 16384);
    wt_pack_kernel<<<64, 256, 0, stream>>>(Bw[l], Kd[l], (l == 3) ? 1 : 100,
                                           wslot + (size_t)(4 + l) * 16384);
  }
  float* outX = (float*)d_out;                 // [N,5]
  float* outB = (float*)d_out + (size_t)N_NODES * 5;  // [N,1]

  // X head
  pack_act_kernel<<<(N_NODES * 128) / 256, 256, 0, stream>>>(hin, actA);
  gemm16_kernel<<<2048, 128, 0, stream>>>(actA, wslot + 0 * 16384, Xb[0], 100, 8, 1, 0, actB, nullptr);
  gemm16_kernel<<<2048, 128, 0, stream>>>(actB, wslot + 1 * 16384, Xb[1], 100, 8, 4, 0, actA, nullptr);
  gemm16_kernel<<<2048, 128, 0, stream>>>(actA, wslot + 2 * 16384, Xb[2], 100, 8, 4, 0, actB, nullptr);
  gemm16_kernel<<< 256, 128, 0, stream>>>(actB, wslot + 3 * 16384, Xb[3],   5, 1, 4, 1, nullptr, outX);

  // B head
  pack_act_kernel<<<(N_NODES * 128) / 256, 256, 0, stream>>>(hin, actA);
  gemm16_kernel<<<2048, 128, 0, stream>>>(actA, wslot + 4 * 16384, Bb[0], 100, 8, 1, 0, actB, nullptr);
  gemm16_kernel<<<2048, 128, 0, stream>>>(actB, wslot + 5 * 16384, Bb[1], 100, 8, 4, 0, actA, nullptr);
  gemm16_kernel<<<2048, 128, 0, stream>>>(actA, wslot + 6 * 16384, Bb[2], 100, 8, 4, 0, actB, nullptr);
  gemm16_kernel<<< 256, 128, 0, stream>>>(actB, wslot + 7 * 16384, Bb[3],   1, 1, 4, 2, nullptr, outB);
}
